// GRUCell_47802986004974
// MI455X (gfx1250) — matrix operand order repaired and checked by reading, not compiled, not run on hardware
//
#include <hip/hip_runtime.h>

#define DI __device__ __forceinline__

typedef __bf16 bf16;
typedef __attribute__((ext_vector_type(16))) __bf16 v16bf;
typedef __attribute__((ext_vector_type(8)))  __bf16 v8bf;
typedef __attribute__((ext_vector_type(8)))  float  v8f;
typedef __attribute__((ext_vector_type(4)))  int    v4i;

#define AS1 __attribute__((address_space(1)))
#define AS3 __attribute__((address_space(3)))

static constexpr int BATCH = 64;
static constexpr int SEQ   = 512;
static constexpr int IN    = 512;
static constexpr int HID   = 1024;
static constexpr int BT    = BATCH * SEQ;   // 32768

// 192KB dynamic LDS: [buf2][ntl2][gate3][kk16][lane32][16 bf16]
static constexpr int    LDS_ELEMS = 2 * 2 * 3 * 16 * 32 * 16;
static constexpr size_t LDS_BYTES = (size_t)LDS_ELEMS * 2;   // 196608

#if defined(__HIP_DEVICE_COMPILE__) && \
    __has_builtin(__builtin_amdgcn_global_load_async_to_lds_b128)
#define HAVE_ASYNC_LDS 1
#endif

// ---------------------------------------------------------------------------
// fp32 -> bf16 conversion (grid-stride)
// ---------------------------------------------------------------------------
__global__ void cvt_bf16_kernel(const float* __restrict__ src,
                                bf16* __restrict__ dst, int n) {
  int i = blockIdx.x * blockDim.x + threadIdx.x;
  int stride = gridDim.x * blockDim.x;
  for (; i < n; i += stride) dst[i] = (bf16)src[i];
}

// ---------------------------------------------------------------------------
// WMMA helpers (CDNA5 layouts, wave32)
// ---------------------------------------------------------------------------
DI v8f wmma_bf16(v16bf a, v16bf b, v8f c) {
  return __builtin_amdgcn_wmma_f32_16x16x32_bf16(
      false, a, false, b, (short)0, c, false, false);
}

// A-matrix 16x32 bf16 tile. Lane l: row = l&15; lanes 0-15 hold
// K in {k0..k0+15}, lanes 16-31 hold {k0+16..k0+31}.
DI v16bf load_a_tile(const bf16* __restrict__ base, int ldk, int lane, int k0) {
  const int row = lane & 15;
  const int hi  = lane >> 4;
  const bf16* p = base + (size_t)row * ldk + k0 + hi * 16;
  v8bf c0 = *(const v8bf*)(p);
  v8bf c1 = *(const v8bf*)(p + 8);
  v16bf a;
#pragma unroll
  for (int i = 0; i < 8; ++i) { a[i] = c0[i]; a[i + 8] = c1[i]; }
  return a;
}

// B-matrix 32x16 bf16 tile, element (k,n) = W[n][k] (W row-major [N][K]).
// Lane l: col = l&15; K = k0 + (l>>4)*16 + e -> one contiguous 32B load.
DI v16bf load_b_tile(const bf16* __restrict__ base, int ldk, int lane, int k0) {
  const int col = lane & 15;
  const int hi  = lane >> 4;
  return *(const v16bf*)(base + (size_t)col * ldk + k0 + hi * 16);
}

DI float sigmoid_(float x) { return 1.f / (1.f + __expf(-x)); }
DI float tanh_(float x) {            // overflow-safe
  float e = __expf(2.f * x);
  return 1.f - 2.f / (e + 1.f);
}

// 16B global -> LDS copy; async path (ASYNCcnt) when the builtin exists.
// Builtin signature (from probe diagnostic): (v4i AS1* src, v4i AS3* dst,
// imm offset, imm cpol).
DI void copy16_to_lds(const bf16* g, bf16* l) {
#if defined(HAVE_ASYNC_LDS)
  bf16* gm = const_cast<bf16*>(g);
  __builtin_amdgcn_global_load_async_to_lds_b128(
      (AS1 v4i*)gm, (AS3 v4i*)l, 0, 0);
#else
  *(v8bf*)l = *(const v8bf*)g;
#endif
}
DI void async_fence_() {
#if defined(HAVE_ASYNC_LDS)
  asm volatile("s_wait_asynccnt 0x0" ::: "memory");
#endif
}

// ---------------------------------------------------------------------------
// Input projections: P_g = X @ W_i_g^T + b_i_g, g in {r,z,n}
// 2 M-tiles per wave (B reused x2), fully double-buffered operand regs.
// grid = (HID/16, BT/32/8), block = 256 (8 waves)
// ---------------------------------------------------------------------------
__global__ void __launch_bounds__(256)
xproj_kernel(const bf16* __restrict__ X,
             const bf16* __restrict__ Wr, const bf16* __restrict__ Wz,
             const bf16* __restrict__ Wn,
             const float* __restrict__ br, const float* __restrict__ bz,
             const float* __restrict__ bn,
             float* __restrict__ Pr, float* __restrict__ Pz,
             float* __restrict__ Pn) {
  const int lane = threadIdx.x & 31;
  const int wave = threadIdx.x >> 5;
  const int mb = blockIdx.y * 8 + wave;   // 0..1023, covers rows mb*32..+31
  const int nt = blockIdx.x;              // 0..63

  const bf16* A0 = X + (size_t)(mb * 2 + 0) * 16 * IN;
  const bf16* A1 = X + (size_t)(mb * 2 + 1) * 16 * IN;
  const bf16* B0 = Wr + (size_t)nt * 16 * IN;
  const bf16* B1 = Wz + (size_t)nt * 16 * IN;
  const bf16* B2 = Wn + (size_t)nt * 16 * IN;

  v8f accr0 = {}, accr1 = {}, accz0 = {}, accz1 = {}, accn0 = {}, accn1 = {};

  v16bf a0 = load_a_tile(A0, IN, lane, 0);
  v16bf a1 = load_a_tile(A1, IN, lane, 0);
  v16bf b0 = load_b_tile(B0, IN, lane, 0);
  v16bf b1 = load_b_tile(B1, IN, lane, 0);
  v16bf b2 = load_b_tile(B2, IN, lane, 0);

#pragma unroll 4
  for (int k0 = 0; k0 < IN; k0 += 32) {
    v16bf na0, na1, nb0, nb1, nb2;
    if (k0 + 32 < IN) {                 // prefetch next k-step
      na0 = load_a_tile(A0, IN, lane, k0 + 32);
      na1 = load_a_tile(A1, IN, lane, k0 + 32);
      nb0 = load_b_tile(B0, IN, lane, k0 + 32);
      nb1 = load_b_tile(B1, IN, lane, k0 + 32);
      nb2 = load_b_tile(B2, IN, lane, k0 + 32);
    }
    accr0 = wmma_bf16(a0, b0, accr0);
    accr1 = wmma_bf16(a1, b0, accr1);
    accz0 = wmma_bf16(a0, b1, accz0);
    accz1 = wmma_bf16(a1, b1, accz1);
    accn0 = wmma_bf16(a0, b2, accn0);
    accn1 = wmma_bf16(a1, b2, accn1);
    if (k0 + 32 < IN) { a0 = na0; a1 = na1; b0 = nb0; b1 = nb1; b2 = nb2; }
  }

  const int hi = lane >> 4;
  const int c  = nt * 16 + (lane & 15);
  const float vbr = br[c], vbz = bz[c], vbn = bn[c];
#pragma unroll
  for (int v = 0; v < 8; ++v) {
    const int m0 = (mb * 2 + 0) * 16 + v + hi * 8;
    const int m1 = (mb * 2 + 1) * 16 + v + hi * 8;
    const size_t o0 = (size_t)m0 * HID + c;
    const size_t o1 = (size_t)m1 * HID + c;
    Pr[o0] = accr0[v] + vbr;  Pr[o1] = accr1[v] + vbr;
    Pz[o0] = accz0[v] + vbz;  Pz[o1] = accz1[v] + vbz;
    Pn[o0] = accn0[v] + vbn;  Pn[o1] = accn1[v] + vbn;
  }
}

// ---------------------------------------------------------------------------
// gru_step LDS staging: layout [buf][ntl][g][kk][lane] x 32B, exactly the
// per-lane WMMA B-operand order, so consumer reads are contiguous 32B ds loads.
// ---------------------------------------------------------------------------
DI void stage_half(int tid, int buf, int half, int nt0,
                   const bf16* __restrict__ Ur, const bf16* __restrict__ Uz,
                   const bf16* __restrict__ Un, bf16* Bsm) {
  const bf16* W[3] = {Ur, Uz, Un};
  // 6144 units of 16B per half; 24 per thread
#pragma unroll 4
  for (int u = tid; u < 6144; u += 256) {
    const int h  = u & 1;
    const int ln = (u >> 1) & 31;
    const int kk = (u >> 6) & 15;
    const int bk = u >> 10;           // 0..5
    const int ntl = bk / 3;
    const int g   = bk - ntl * 3;
    const bf16* gsrc = W[g] +
        (size_t)((nt0 + ntl) * 16 + (ln & 15)) * HID +
        half * 512 + kk * 32 + (ln >> 4) * 16 + h * 8;
    bf16* ldst = Bsm +
        ((size_t)(((buf * 2 + ntl) * 3 + g) * 16 + kk) * 32 + ln) * 16 + h * 8;
    copy16_to_lds(gsrc, ldst);
  }
}

DI v16bf lds_b(const bf16* Bsm, int buf, int ntl, int g, int kk, int lane) {
  return *(const v16bf*)(
      Bsm + ((size_t)(((buf * 2 + ntl) * 3 + g) * 16 + kk) * 32 + lane) * 16);
}

DI void compute_half(v8f& accr, v8f& accz, v8f& accn,
                     const bf16* __restrict__ Abase, const bf16* Bsm,
                     int buf, int ntl, int half, int lane) {
  v16bf a = load_a_tile(Abase, HID, lane, half * 512);
#pragma unroll 4
  for (int kk = 0; kk < 16; ++kk) {
    v16bf an = a;
    if (kk < 15) an = load_a_tile(Abase, HID, lane, half * 512 + (kk + 1) * 32);
    v16bf br = lds_b(Bsm, buf, ntl, 0, kk, lane);
    v16bf bz = lds_b(Bsm, buf, ntl, 1, kk, lane);
    v16bf bn = lds_b(Bsm, buf, ntl, 2, kk, lane);
    accr = wmma_bf16(a, br, accr);
    accz = wmma_bf16(a, bz, accz);
    accn = wmma_bf16(a, bn, accn);
    a = an;
  }
}

// ---------------------------------------------------------------------------
// One GRU time step, all three gates fused per 16x16 tile:
//   r = sigmoid(Pr + h@Ur^T + bhr); z = sigmoid(Pz + h@Uz^T + bhz)
//   n = tanh(Pn + r*(h@Un^T + bhn)); h' = (1-z)*n + z*h
// 32 blocks x 8 waves; WG covers 2 n-tiles x 4 m-tiles. Recurrent weights
// staged to LDS (K split in halves, async double-buffered against WMMAs).
// ---------------------------------------------------------------------------
__global__ void __launch_bounds__(256)
gru_step_kernel(int t,
                const float* __restrict__ Pr, const float* __restrict__ Pz,
                const float* __restrict__ Pn,
                const bf16* __restrict__ Ur, const bf16* __restrict__ Uz,
                const bf16* __restrict__ Un,
                const float* __restrict__ bhr, const float* __restrict__ bhz,
                const float* __restrict__ bhn,
                const bf16* __restrict__ h_in_b, bf16* __restrict__ h_out_b,
                const float* __restrict__ h_in_f, float* __restrict__ h_out_f,
                float* __restrict__ out) {
  extern __shared__ bf16 Bsm[];
  const int tid  = threadIdx.x;
  const int lane = tid & 31;
  const int wave = tid >> 5;
  const int mt  = wave & 3;            // batch tile 0..3
  const int ntl = wave >> 2;           // 0..1
  const int nt0 = blockIdx.x * 2;
  const int nt  = nt0 + ntl;           // 0..63
  const int hi  = lane >> 4;
  const int c   = nt * 16 + (lane & 15);

  // stage first K-half for this WG's two n-tiles
  stage_half(tid, 0, 0, nt0, Ur, Uz, Un, Bsm);

  // seed r/z accumulators with the precomputed input projections
  v8f accr, accz, accn;
#pragma unroll
  for (int v = 0; v < 8; ++v) {
    const int m = mt * 16 + v + hi * 8;                 // batch index
    const size_t o = ((size_t)m * SEQ + t) * HID + c;   // [B,T,H]
    accr[v] = Pr[o];
    accz[v] = Pz[o];
    accn[v] = 0.f;
  }

  async_fence_();
  __syncthreads();

  // stage second half while first half's WMMAs run
  stage_half(tid, 1, 1, nt0, Ur, Uz, Un, Bsm);

  const bf16* Abase = h_in_b + (size_t)mt * 16 * HID;
  compute_half(accr, accz, accn, Abase, Bsm, 0, ntl, 0, lane);

  async_fence_();
  __syncthreads();

  compute_half(accr, accz, accn, Abase, Bsm, 1, ntl, 1, lane);

  const float vbr = bhr[c], vbz = bhz[c], vbn = bhn[c];
#pragma unroll
  for (int v = 0; v < 8; ++v) {
    const int m = mt * 16 + v + hi * 8;
    const size_t oh = (size_t)m * HID + c;
    const size_t ot = ((size_t)m * SEQ + t) * HID + c;
    const float r = sigmoid_(accr[v] + vbr);
    const float z = sigmoid_(accz[v] + vbz);
    const float n = tanh_(Pn[ot] + r * (accn[v] + vbn));
    const float hprev = h_in_f[oh];
    const float hnew = (1.f - z) * n + z * hprev;
    out[ot]     = hnew;
    h_out_f[oh] = hnew;
    h_out_b[oh] = (bf16)hnew;
  }
}

// ---------------------------------------------------------------------------
// host side
// ---------------------------------------------------------------------------
extern "C" void kernel_launch(void* const* d_in, const int* in_sizes, int n_in,
                              void* d_out, int out_size, void* d_ws,
                              size_t ws_size, hipStream_t stream) {
  const float* x   = (const float*)d_in[0];
  const float* Wir = (const float*)d_in[1];
  const float* bir = (const float*)d_in[2];
  const float* Whr = (const float*)d_in[3];
  const float* bhr = (const float*)d_in[4];
  const float* Wiz = (const float*)d_in[5];
  const float* biz = (const float*)d_in[6];
  const float* Whz = (const float*)d_in[7];
  const float* bhz = (const float*)d_in[8];
  const float* Win = (const float*)d_in[9];
  const float* bin = (const float*)d_in[10];
  const float* Whn = (const float*)d_in[11];
  const float* bhn = (const float*)d_in[12];
  float* out = (float*)d_out;

  char* ws = (char*)d_ws;
  size_t off = 0;
  auto carve = [&](size_t bytes) -> char* {
    char* p = ws + off;
    off = (off + bytes + 255) & ~(size_t)255;
    return p;
  };
  bf16* Xb   = (bf16*)carve((size_t)BT * IN * 2);
  bf16* Wirb = (bf16*)carve((size_t)HID * IN * 2);
  bf16* Wizb = (bf16*)carve((size_t)HID * IN * 2);
  bf16* Winb = (bf16*)carve((size_t)HID * IN * 2);
  bf16* Whrb = (bf16*)carve((size_t)HID * HID * 2);
  bf16* Whzb = (bf16*)carve((size_t)HID * HID * 2);
  bf16* Whnb = (bf16*)carve((size_t)HID * HID * 2);
  float* Pr  = (float*)carve((size_t)BT * HID * 4);
  float* Pz  = (float*)carve((size_t)BT * HID * 4);
  float* Pn  = (float*)carve((size_t)BT * HID * 4);
  bf16* hb0  = (bf16*)carve((size_t)BATCH * HID * 2);
  bf16* hb1  = (bf16*)carve((size_t)BATCH * HID * 2);
  float* hf0 = (float*)carve((size_t)BATCH * HID * 4);
  float* hf1 = (float*)carve((size_t)BATCH * HID * 4);
  bf16* hb[2] = {hb0, hb1};
  float* hf[2] = {hf0, hf1};

  static_cast<void>(hipMemsetAsync(hb0, 0, (size_t)BATCH * HID * 2, stream));
  static_cast<void>(hipMemsetAsync(hf0, 0, (size_t)BATCH * HID * 4, stream));

  auto cvt = [&](const float* s, bf16* d, int n) {
    int blocks = (n + 256 * 8 - 1) / (256 * 8);
    if (blocks > 4096) blocks = 4096;
    cvt_bf16_kernel<<<blocks, 256, 0, stream>>>(s, d, n);
  };
  cvt(x,   Xb,   BT * IN);
  cvt(Wir, Wirb, HID * IN);
  cvt(Wiz, Wizb, HID * IN);
  cvt(Win, Winb, HID * IN);
  cvt(Whr, Whrb, HID * HID);
  cvt(Whz, Whzb, HID * HID);
  cvt(Whn, Whnb, HID * HID);

  dim3 pgrid(HID / 16, BT / 32 / 8);   // (64, 128)
  xproj_kernel<<<pgrid, 256, 0, stream>>>(Xb, Wirb, Wizb, Winb, bir, biz, bin,
                                          Pr, Pz, Pn);

  // allow 192KB dynamic LDS (host-side attribute set; capture-safe)
  static_cast<void>(hipFuncSetAttribute(
      reinterpret_cast<const void*>(gru_step_kernel),
      hipFuncAttributeMaxDynamicSharedMemorySize, (int)LDS_BYTES));

  for (int t = 0; t < SEQ; ++t) {
    gru_step_kernel<<<32, 256, LDS_BYTES, stream>>>(
        t, Pr, Pz, Pn, Whrb, Whzb, Whnb, bhr, bhz, bhn,
        hb[t & 1], hb[(t + 1) & 1], hf[t & 1], hf[(t + 1) & 1], out);
  }
}
